// SSPFM_24713241821851
// MI455X (gfx1250) — matrix operand order repaired and checked by reading, not compiled, not run on hardware
//
#include <hip/hip_runtime.h>
#include <hip/hip_bf16.h>

typedef __attribute__((ext_vector_type(16))) _Float16 v16h;
typedef __attribute__((ext_vector_type(8)))  _Float16 v8h;
typedef __attribute__((ext_vector_type(8)))  float    v8f;
typedef __attribute__((ext_vector_type(4)))  float    v4f;
typedef __attribute__((ext_vector_type(2)))  float    v2f;

#define B_ 4
#define C_ 256
#define N_ 4096          // 64*64
#define BN_ (B_ * N_)    // 16384
#define TN 64            // pixel tile per block
#define RS 272           // LDS row stride in halves (32B aligned rows)

// ---------------------------------------------------------------------------
// Kernel 0: convert fp32 weights -> f16, packed [Wv | Wk | Wq] (conv loop order)
// ---------------------------------------------------------------------------
__global__ __launch_bounds__(256) void k_cvt_weights(
    const float* __restrict__ Wq, const float* __restrict__ Wk,
    const float* __restrict__ Wv, _Float16* __restrict__ W16)
{
    int idx = blockIdx.x * 256 + threadIdx.x;          // 3*65536 total
    int s   = idx >> 16;
    int off = idx & 65535;
    const float* src = (s == 0) ? Wv : ((s == 1) ? Wk : Wq);
    W16[idx] = (_Float16)src[off];
}

// ---------------------------------------------------------------------------
// Kernel 1: fused 1x1-conv (WMMA GEMM) + channel max/mean for v, k, q.
// Block = 128 threads (4 waves), each wave owns 16 pixels of a 64-pixel tile.
// Writes packed (max, mean) float2 per pixel per conv.
// ---------------------------------------------------------------------------
__global__ __launch_bounds__(128) void k_conv_reduce(
    const float* __restrict__ x,          // [B, C, N]
    const _Float16* __restrict__ W16,     // [3][C][C]  (v,k,q)
    const float* __restrict__ bv, const float* __restrict__ bk,
    const float* __restrict__ bq,
    v2f* __restrict__ apk,                // [B*N] (a_max, a_mean)
    v2f* __restrict__ kpk,                // [B*N] (k_max, k_mean)
    v2f* __restrict__ qpk)                // [B*N] (q_max, q_mean)
{
    __shared__ __align__(32) _Float16 xs[TN * RS];

    const int tid  = threadIdx.x;
    const int tile = blockIdx.x & 63;     // N_/TN = 64 tiles
    const int b    = blockIdx.x >> 6;
    const int n0   = tile * TN;

    // Stage x tile into LDS as f16, layout [pixel][channel] (row stride RS).
    // x is read exactly once globally -> non-temporal loads.
    for (int idx = tid; idx < (C_ * TN) / 4; idx += 128) {
        int n4 = idx & 15;                // 16 float4 per channel row
        int c  = idx >> 4;
        const v4f v = __builtin_nontemporal_load(
            (const v4f*)(x + ((size_t)b * C_ + c) * N_ + n0 + n4 * 4));
        int base = (n4 * 4) * RS + c;
        xs[base         ] = (_Float16)v.x;
        xs[base +     RS] = (_Float16)v.y;
        xs[base + 2 * RS] = (_Float16)v.z;
        xs[base + 3 * RS] = (_Float16)v.w;
    }
    __syncthreads();

    const int lane = tid & 31;
    const int wv   = tid >> 5;            // wave 0..3
    const int col  = lane & 15;           // N index within wave tile / A row M%16
    const int h    = lane >> 4;           // half-wave selector
    const _Float16* xcol = xs + (wv * 16 + col) * RS;

    for (int conv = 0; conv < 3; ++conv) {
        const _Float16* Wp   = W16 + (size_t)conv * C_ * C_;
        const float*    bias = (conv == 0) ? bv : ((conv == 1) ? bk : bq);

        float gmax = -3.0e38f;
        float gsum = 0.0f;

        for (int m = 0; m < 16; ++m) {                     // 16 output-channel tiles
            const int mrow = m * 16 + col;                 // A-matrix row for this lane
            const _Float16* wrow = Wp + (size_t)mrow * C_;

            v8f acc;
            #pragma unroll
            for (int r = 0; r < 8; ++r)                    // fold bias into C
                acc[r] = bias[m * 16 + r + 8 * h];

            #pragma unroll
            for (int k0 = 0; k0 < C_; k0 += 32) {
                // A 16x32 f16: halves 0-15 = K {16h..16h+15}
                v16h a  = *(const v16h*)(wrow + k0 + 16 * h);
                // B 32x16 f16: lane=col holds K {16h..16h+15}, one 32B LDS read
                v16h bm = *(const v16h*)(xcol + k0 + 16 * h);
                acc = __builtin_amdgcn_wmma_f32_16x16x32_f16(
                          false, a, false, bm, (short)0, acc, false, false);
            }

            // reduce this 16-row tile: 8 rows in-lane, 8 more in lane^16
            float tmax = acc[0], tsum = acc[0];
            #pragma unroll
            for (int r = 1; r < 8; ++r) { tmax = fmaxf(tmax, acc[r]); tsum += acc[r]; }
            float omx = __shfl_xor(tmax, 16);
            float osm = __shfl_xor(tsum, 16);
            tmax = fmaxf(tmax, omx);
            tsum += osm;
            gmax = fmaxf(gmax, tmax);
            gsum += tsum;
        }

        v2f* dst = (conv == 0) ? apk : ((conv == 1) ? kpk : qpk);
        if (h == 0) {                                      // lanes 0-15: one pixel each
            int gi = b * N_ + n0 + wv * 16 + col;
            v2f o;
            o.x = gmax;
            o.y = gsum * (1.0f / 256.0f);
            dst[gi] = o;
        }
    }
}

// ---------------------------------------------------------------------------
// Kernel 2: Z[i] = sum_j exp(qm[i]*km[j] + qn[i]*kn[j]);
//           w[i] = (alpha*am[i] + beta*an[i]) / Z[i]
// Emits packed (qm, qn, w, 0) for kernel 3 (one b128 LDS read per exp there).
// ---------------------------------------------------------------------------
__global__ __launch_bounds__(256) void k_softmax_z(
    const v2f* __restrict__ qpk, const v2f* __restrict__ kpk,
    const v2f* __restrict__ apk,
    const float* __restrict__ alpha, const float* __restrict__ beta,
    v4f* __restrict__ qw)
{
    __shared__ v2f sk[1024];
    const int i = blockIdx.x * 256 + threadIdx.x;
    const int b = i >> 12;                                 // same for whole block
    const v2f q2 = qpk[i];
    const v2f a2 = apk[i];
    const v2f* kb = kpk + (b << 12);

    float z = 0.0f;
    for (int ch = 0; ch < N_; ch += 1024) {
        __syncthreads();
        for (int t = threadIdx.x; t < 1024; t += 256)
            sk[t] = kb[ch + t];
        __syncthreads();
        #pragma unroll 8
        for (int j = 0; j < 1024; ++j) {
            v2f k2 = sk[j];                                // one ds_load_b64
            z += __expf(fmaf(q2.x, k2.x, q2.y * k2.y));
        }
    }
    float w = (alpha[0] * a2.x + beta[0] * a2.y) / z;
    v4f o;
    o.x = q2.x; o.y = q2.y; o.z = w; o.w = 0.0f;
    qw[i] = o;
}

// ---------------------------------------------------------------------------
// Kernel 3: kk[j] = sum_i w[i] * exp(qm[i]*km[j] + qn[i]*kn[j])
// ---------------------------------------------------------------------------
__global__ __launch_bounds__(256) void k_attn_out(
    const v4f* __restrict__ qw, const v2f* __restrict__ kpk,
    float* __restrict__ kk)
{
    __shared__ v4f sq[1024];
    const int j = blockIdx.x * 256 + threadIdx.x;
    const int b = j >> 12;
    const v2f k2 = kpk[j];
    const v4f* qb = qw + (b << 12);

    float acc = 0.0f;
    for (int ch = 0; ch < N_; ch += 1024) {
        __syncthreads();
        for (int t = threadIdx.x; t < 1024; t += 256)
            sq[t] = qb[ch + t];
        __syncthreads();
        #pragma unroll 8
        for (int i = 0; i < 1024; ++i) {
            v4f q4 = sq[i];                                // one ds_load_b128
            acc = fmaf(q4.z, __expf(fmaf(q4.x, k2.x, q4.y * k2.y)), acc);
        }
    }
    kk[j] = acc;
}

// ---------------------------------------------------------------------------
// Kernel 4: broadcast kk[b,n] across the C channel dim, non-temporal b128 stores.
// ---------------------------------------------------------------------------
__global__ __launch_bounds__(256) void k_broadcast(
    const float* __restrict__ kk, float* __restrict__ out)
{
    const int idx = blockIdx.x * 256 + threadIdx.x;        // float4 index
    const int n4  = idx & 1023;                            // N_/4
    const int b   = idx >> 18;                             // / (C_ * N_/4)
    const v4f v = ((const v4f*)(kk + (b << 12)))[n4];
    __builtin_nontemporal_store(v, ((v4f*)out) + idx);
}

// ---------------------------------------------------------------------------
extern "C" void kernel_launch(void* const* d_in, const int* in_sizes, int n_in,
                              void* d_out, int out_size, void* d_ws, size_t ws_size,
                              hipStream_t stream) {
    const float* x     = (const float*)d_in[0];
    const float* Wq    = (const float*)d_in[1];
    const float* bq    = (const float*)d_in[2];
    const float* Wk    = (const float*)d_in[3];
    const float* bk    = (const float*)d_in[4];
    const float* Wv    = (const float*)d_in[5];
    const float* bv    = (const float*)d_in[6];
    const float* alpha = (const float*)d_in[7];
    const float* beta  = (const float*)d_in[8];

    // workspace layout (all 16B aligned)
    char* p = (char*)d_ws;
    _Float16* W16 = (_Float16*)p;  p += (size_t)3 * C_ * C_ * sizeof(_Float16); // 384 KB
    v2f* apk = (v2f*)p;            p += (size_t)BN_ * sizeof(v2f);              // 128 KB
    v2f* kpk = (v2f*)p;            p += (size_t)BN_ * sizeof(v2f);              // 128 KB
    v2f* qpk = (v2f*)p;            p += (size_t)BN_ * sizeof(v2f);              // 128 KB
    v4f* qw  = (v4f*)p;            p += (size_t)BN_ * sizeof(v4f);              // 256 KB
    float* kk = (float*)p;                                                      //  64 KB

    k_cvt_weights<<<(3 * C_ * C_) / 256, 256, 0, stream>>>(Wq, Wk, Wv, W16);

    k_conv_reduce<<<B_ * (N_ / TN), 128, 0, stream>>>(x, W16, bv, bk, bq,
                                                      apk, kpk, qpk);

    k_softmax_z<<<BN_ / 256, 256, 0, stream>>>(qpk, kpk, apk, alpha, beta, qw);

    k_attn_out<<<BN_ / 256, 256, 0, stream>>>(qw, kpk, kk);

    k_broadcast<<<(B_ * C_ * N_ / 4) / 256, 256, 0, stream>>>(kk, (float*)d_out);
}
